// ScaledDotProductAttention_61607010894461
// MI455X (gfx1250) — matrix operand order repaired and checked by reading, not compiled, not run on hardware
//
#include <hip/hip_runtime.h>

// ---------------------------------------------------------------------------
// Causal flash-attention forward, B=4 H=16 S=2048 D=64 (fp32 in/out), gfx1250.
// v_wmma_f32_16x16x32_f16 for QK^T and PV, online softmax with v_permlane16
// butterfly reductions (no LDS round-trips), double-buffered K/V LDS tiles,
// packed f32->f16 conversions. Wave32; A/B k order contiguous (k = 16*half + s).
// ---------------------------------------------------------------------------

typedef __attribute__((ext_vector_type(16))) _Float16 v16h;
typedef __attribute__((ext_vector_type(8)))  _Float16 v8h;
typedef __attribute__((ext_vector_type(4)))  _Float16 v4h;
typedef __attribute__((ext_vector_type(2)))  __fp16   v2fp;   // cvt_pkrtz result type
typedef __attribute__((ext_vector_type(8)))  float    v8f;

#define BB 4
#define HH 16
#define SS 2048
#define DD 64

#define BR 64          // query rows per workgroup
#define BC 64          // key/value columns per iteration
#define NW 4           // waves per workgroup (wave32)
#define LSTR 80        // LDS row stride in halves (160B -> 32B-aligned frags)

union V16U { v16h v; v8h h[2]; };
union H4U  { v4h v; v2fp p[2]; };
union H16U { v16h v; v2fp p[8]; };

// ---- cross-lane butterfly (xor) within each 16-lane half, VALU-only ------
#if __has_builtin(__builtin_amdgcn_permlane16)
static __device__ __forceinline__ float lanexor16(float v, unsigned s0, unsigned s1) {
    union { float f; unsigned u; } a, b;
    a.f = v;
    b.u = __builtin_amdgcn_permlane16(a.u, a.u, s0, s1, false, false);
    return b.f;
}
#define XOR1(x)  lanexor16(x, 0x67452301u, 0xEFCDAB89u)
#define XOR2(x)  lanexor16(x, 0x45670123u, 0xCDEF89ABu)
#define XOR4(x)  lanexor16(x, 0x32107654u, 0xBA98FEDCu)
#define XOR8(x)  lanexor16(x, 0xFEDCBA98u, 0x76543210u)
#else
#define XOR1(x)  __shfl_xor((x), 1, 32)
#define XOR2(x)  __shfl_xor((x), 2, 32)
#define XOR4(x)  __shfl_xor((x), 4, 32)
#define XOR8(x)  __shfl_xor((x), 8, 32)
#endif

// ---- cooperative global->LDS tile fill (f32 -> f16) ----------------------
static __device__ __forceinline__
void load_tiles(const float* __restrict__ Kb, const float* __restrict__ Vb,
                int kb, _Float16* __restrict__ ksm, _Float16* __restrict__ vsm,
                int tid)
{
#pragma unroll
    for (int it = 0; it < (BC * (DD / 4)) / (NW * 32); ++it) {
        int idx = tid + it * (NW * 32);
        int row = idx >> 4, seg = idx & 15;
        // K tile: row-major [n][d]
        float4 fk = *(const float4*)(Kb + (size_t)(kb * BC + row) * DD + seg * 4);
        H4U hk;
        hk.p[0] = __builtin_amdgcn_cvt_pkrtz(fk.x, fk.y);
        hk.p[1] = __builtin_amdgcn_cvt_pkrtz(fk.z, fk.w);
        *(v4h*)(&ksm[row * LSTR + seg * 4]) = hk.v;
        // V tile: transposed [d][kc]
        float4 fv = *(const float4*)(Vb + (size_t)(kb * BC + row) * DD + seg * 4);
        vsm[(seg * 4 + 0) * LSTR + row] = (_Float16)fv.x;
        vsm[(seg * 4 + 1) * LSTR + row] = (_Float16)fv.y;
        vsm[(seg * 4 + 2) * LSTR + row] = (_Float16)fv.z;
        vsm[(seg * 4 + 3) * LSTR + row] = (_Float16)fv.w;
    }
}

__global__ __launch_bounds__(NW * 32)
void fa2_causal_wmma_kernel(const float* __restrict__ Q,
                            const float* __restrict__ K,
                            const float* __restrict__ V,
                            float* __restrict__ O)
{
    __shared__ __align__(16) _Float16 Ks [2][BC * LSTR];       // K tiles, [n][d]
    __shared__ __align__(16) _Float16 Vts[2][DD * LSTR];       // V tiles transposed
    __shared__ __align__(16) _Float16 Ps [NW * 16 * LSTR];     // per-wave P slab

    const int tid  = threadIdx.x;
    const int wave = tid >> 5;
    const int lane = tid & 31;
    const int g    = lane >> 4;     // half-wave group (0/1)
    const int nn   = lane & 15;     // N / column-in-tile lane index
    const int qb   = blockIdx.x;    // 64-row query block
    const int bh   = blockIdx.y;    // fused batch*head

    const float scale = 0.125f;     // 1/sqrt(64)

    const float* Qb = Q + (size_t)bh * SS * DD;
    const float* Kb = K + (size_t)bh * SS * DD;
    const float* Vb = V + (size_t)bh * SS * DD;
    float*       Ob = O + (size_t)bh * SS * DD;

    // ---- Q fragment: A-matrix layout (lane = M, K striped), prescaled ----
    const int qrow = qb * BR + wave * 16 + nn;              // M = lane%16
    v16h aq[2];
#pragma unroll
    for (int c = 0; c < 2; ++c) {
        const float* qp = Qb + (size_t)qrow * DD + c * 32 + g * 16;
        float4 f0 = *(const float4*)(qp + 0);
        float4 f1 = *(const float4*)(qp + 4);
        float4 f2 = *(const float4*)(qp + 8);
        float4 f3 = *(const float4*)(qp + 12);
        H16U u;
        u.p[0] = __builtin_amdgcn_cvt_pkrtz(f0.x * scale, f0.y * scale);
        u.p[1] = __builtin_amdgcn_cvt_pkrtz(f0.z * scale, f0.w * scale);
        u.p[2] = __builtin_amdgcn_cvt_pkrtz(f1.x * scale, f1.y * scale);
        u.p[3] = __builtin_amdgcn_cvt_pkrtz(f1.z * scale, f1.w * scale);
        u.p[4] = __builtin_amdgcn_cvt_pkrtz(f2.x * scale, f2.y * scale);
        u.p[5] = __builtin_amdgcn_cvt_pkrtz(f2.z * scale, f2.w * scale);
        u.p[6] = __builtin_amdgcn_cvt_pkrtz(f3.x * scale, f3.y * scale);
        u.p[7] = __builtin_amdgcn_cvt_pkrtz(f3.z * scale, f3.w * scale);
        aq[c] = u.v;
    }

    // ---- running softmax state + output accumulators ----
    float m_i[8], l_i[8];
    v8f acc[4];
#pragma unroll
    for (int r = 0; r < 8; ++r) { m_i[r] = -__builtin_inff(); l_i[r] = 0.0f; }
#pragma unroll
    for (int t = 0; t < 4; ++t)
#pragma unroll
        for (int r = 0; r < 8; ++r) acc[t][r] = 0.0f;

    // ---- prologue: fill buffer 0 ----
    load_tiles(Kb, Vb, 0, Ks[0], Vts[0], tid);
    __syncthreads();

    for (int kb = 0; kb <= qb; ++kb) {
        const int p = kb & 1;

        // ---- issue global loads of the NEXT tile into the other buffer ----
        if (kb < qb)
            load_tiles(Kb, Vb, kb + 1, Ks[p ^ 1], Vts[p ^ 1], tid);

        const _Float16* ksm = Ks[p];
        const _Float16* vsm = Vts[p];

        // ---- preload all K B-fragments, then the WMMA chain ----
        v16h bk[4][2];
#pragma unroll
        for (int t = 0; t < 4; ++t)
#pragma unroll
            for (int c = 0; c < 2; ++c)
                bk[t][c] = *(const v16h*)(&ksm[(t * 16 + nn) * LSTR + c * 32 + g * 16]);

        v8f s[4];
#pragma unroll
        for (int t = 0; t < 4; ++t) {
            v8f c0;
#pragma unroll
            for (int r = 0; r < 8; ++r) c0[r] = 0.0f;
            c0 = __builtin_amdgcn_wmma_f32_16x16x32_f16(
                    false, aq[0], false, bk[t][0], (short)0, c0, false, false);
            c0 = __builtin_amdgcn_wmma_f32_16x16x32_f16(
                    false, aq[1], false, bk[t][1], (short)0, c0, false, false);
            s[t] = c0;
        }

        // ---- causal mask (diagonal block only) ----
        if (kb == qb) {
#pragma unroll
            for (int t = 0; t < 4; ++t) {
                int colg = t * 16 + nn;
#pragma unroll
                for (int r = 0; r < 8; ++r) {
                    int rowg = wave * 16 + 8 * g + r;
                    if (colg > rowg) s[t][r] = -__builtin_inff();
                }
            }
        }

        // ---- online softmax: VALU butterfly reductions over 16 lanes ----
        float mx[8];
#pragma unroll
        for (int r = 0; r < 8; ++r)
            mx[r] = fmaxf(fmaxf(s[0][r], s[1][r]), fmaxf(s[2][r], s[3][r]));
#pragma unroll
        for (int r = 0; r < 8; ++r) mx[r] = fmaxf(mx[r], XOR1(mx[r]));
#pragma unroll
        for (int r = 0; r < 8; ++r) mx[r] = fmaxf(mx[r], XOR2(mx[r]));
#pragma unroll
        for (int r = 0; r < 8; ++r) mx[r] = fmaxf(mx[r], XOR4(mx[r]));
#pragma unroll
        for (int r = 0; r < 8; ++r) mx[r] = fmaxf(mx[r], XOR8(mx[r]));

        float alpha[8], rs[8];
#pragma unroll
        for (int r = 0; r < 8; ++r) {
            float mnew = fmaxf(m_i[r], mx[r]);
            alpha[r] = __expf(m_i[r] - mnew);
            m_i[r] = mnew;
            float acc_s = 0.0f;
#pragma unroll
            for (int t = 0; t < 4; ++t) {
                float pe = __expf(s[t][r] - mnew);
                s[t][r] = pe;
                acc_s += pe;
            }
            rs[r] = acc_s;
        }
#pragma unroll
        for (int r = 0; r < 8; ++r) rs[r] += XOR1(rs[r]);
#pragma unroll
        for (int r = 0; r < 8; ++r) rs[r] += XOR2(rs[r]);
#pragma unroll
        for (int r = 0; r < 8; ++r) rs[r] += XOR4(rs[r]);
#pragma unroll
        for (int r = 0; r < 8; ++r) rs[r] += XOR8(rs[r]);
#pragma unroll
        for (int r = 0; r < 8; ++r) {
            l_i[r] = l_i[r] * alpha[r] + rs[r];
#pragma unroll
            for (int t = 0; t < 4; ++t) acc[t][r] *= alpha[r];
        }

        // ---- re-layout P: C-layout regs -> per-wave LDS slab -> A-layout ----
        _Float16* pw = &Ps[wave * 16 * LSTR];
#pragma unroll
        for (int t = 0; t < 4; ++t)
#pragma unroll
            for (int r = 0; r < 8; ++r)
                pw[(8 * g + r) * LSTR + t * 16 + nn] = (_Float16)s[t][r];
        // same-wave LDS is in-order: no barrier needed before readback
        v16h ap[2];
#pragma unroll
        for (int c = 0; c < 2; ++c) {
            const _Float16* p0 = &pw[nn * LSTR + c * 32 + g * 16];
            V16U u;
            u.h[0] = *(const v8h*)(p0);
            u.h[1] = *(const v8h*)(p0 + 8);
            ap[c] = u.v;
        }

        // ---- preload all V B-fragments, then the PV WMMA chain ----
        v16h bv[4][2];
#pragma unroll
        for (int t = 0; t < 4; ++t)
#pragma unroll
            for (int c = 0; c < 2; ++c)
                bv[t][c] = *(const v16h*)(&vsm[(t * 16 + nn) * LSTR + c * 32 + g * 16]);
#pragma unroll
        for (int t = 0; t < 4; ++t) {
            acc[t] = __builtin_amdgcn_wmma_f32_16x16x32_f16(
                        false, ap[0], false, bv[t][0], (short)0, acc[t], false, false);
            acc[t] = __builtin_amdgcn_wmma_f32_16x16x32_f16(
                        false, ap[1], false, bv[t][1], (short)0, acc[t], false, false);
        }

        // next iteration overwrites the buffer everyone just read: barrier
        __syncthreads();
    }

    // ---- epilogue: O = acc / l ----
#pragma unroll
    for (int r = 0; r < 8; ++r) {
        float inv = 1.0f / l_i[r];
        int row = qb * BR + wave * 16 + 8 * g + r;
        float* op = Ob + (size_t)row * DD;
#pragma unroll
        for (int t = 0; t < 4; ++t)
            op[t * 16 + nn] = acc[t][r] * inv;
    }
}

extern "C" void kernel_launch(void* const* d_in, const int* in_sizes, int n_in,
                              void* d_out, int out_size, void* d_ws, size_t ws_size,
                              hipStream_t stream) {
    (void)in_sizes; (void)n_in; (void)d_ws; (void)ws_size; (void)out_size;
    const float* Q = (const float*)d_in[0];
    const float* K = (const float*)d_in[1];
    const float* V = (const float*)d_in[2];
    // d_in[3] is the causal mask; causality is applied analytically.
    float* O = (float*)d_out;

    dim3 grid(SS / BR, BB * HH);
    dim3 block(NW * 32);
    fa2_causal_wmma_kernel<<<grid, block, 0, stream>>>(Q, K, V, O);
}
